// GAT_57612691309076
// MI455X (gfx1250) — matrix operand order repaired and checked by reading, not compiled, not run on hardware
//
#include <hip/hip_runtime.h>
#include <math.h>

// ---------------------------------------------------------------------------
// GAT forward for MI455X (gfx1250, wave32, WMMA).
// B=4, N=2048, in=16, hid=32, out=16, heads=6.
// Flash-style fused attention per 16-row tile:
//   pass A : branchless masked row-max sweep (v_cndmask + v_max only)
//   pass C : single WMMA sweep that (a) rebuilds the 16x32 P tile in f16
//            A-fragment layout with branchless exp (select on the argument,
//            exp(-inf)=0 handles the mask), (b) accumulates the per-lane
//            partial softmax denominator, and (c) issues
//            v_wmma_f32_16x16x32_f16 against pre-transposed f16 h tiles.
// Epilogues are fully branchless (rcp for 1/l, max/min identity for ELU).
// ---------------------------------------------------------------------------

typedef __attribute__((ext_vector_type(16))) _Float16 v16h;
typedef __attribute__((ext_vector_type(8)))  _Float16 v8h;
typedef __attribute__((ext_vector_type(8)))  float    v8f;

union V16H { v16h v; _Float16 h[16]; };
union V8F  { v8f  v; float    f[8];  };

#define LRELU_ALPHA 0.2f

constexpr int B_   = 4;
constexpr int N_   = 2048;
constexpr int FIN  = 16;
constexpr int HID  = 32;
constexpr int OUTC = 16;
constexpr int NH   = 6;
constexpr int CAT  = HID * NH;   // 192

__device__ __forceinline__ void load8f(const float* __restrict__ p, float* v) {
    float4 u0 = *(const float4*)p;
    float4 u1 = *(const float4*)(p + 4);
    v[0] = u0.x; v[1] = u0.y; v[2] = u0.z; v[3] = u0.w;
    v[4] = u1.x; v[5] = u1.y; v[6] = u1.z; v[7] = u1.w;
}

// Branchless ELU: exact for y>=0 (exp(0)-1 == 0), straight-line codegen.
__device__ __forceinline__ float elu_bl(float y) {
    return fmaxf(y, 0.f) + (__expf(fminf(y, 0.f)) - 1.f);
}

// ---------------------------------------------------------------------------
// K1: layer-1 projections.  h1 = x @ W1[h]  -> stored transposed f16 [bh][o][n]
//     s1 = h1 . a1[:HID],  s2 = h1 . a1[HID:]
// ---------------------------------------------------------------------------
__global__ __launch_bounds__(256)
void k1_proj1(const float* __restrict__ x, const float* __restrict__ W1,
              const float* __restrict__ a1, _Float16* __restrict__ h1T,
              float* __restrict__ s1, float* __restrict__ s2) {
    __shared__ float Wl[FIN * HID];
    __shared__ float al[2 * HID];
    int idx = blockIdx.x * 256 + threadIdx.x;   // ((b*NH + h) * N + n)
    int n   = idx & (N_ - 1);
    int bh  = idx >> 11;                        // N_ == 2048
    int h   = bh % NH;
    for (int t = threadIdx.x; t < FIN * HID; t += 256)
        Wl[t] = W1[h * FIN * HID + t];
    if (threadIdx.x < 2 * HID)
        al[threadIdx.x] = a1[h * 2 * HID + threadIdx.x];
    __syncthreads();

    float xr[FIN];
    const float* xp = x + ((size_t)(bh / NH) * N_ + n) * FIN;
    #pragma unroll
    for (int k = 0; k < FIN; ++k) xr[k] = xp[k];

    float sa = 0.f, sb = 0.f;
    size_t base = (size_t)bh * HID * N_;
    #pragma unroll
    for (int o = 0; o < HID; ++o) {
        float acc = 0.f;
        #pragma unroll
        for (int k = 0; k < FIN; ++k) acc = fmaf(xr[k], Wl[k * HID + o], acc);
        h1T[base + (size_t)o * N_ + n] = (_Float16)acc;
        sa = fmaf(acc, al[o], sa);
        sb = fmaf(acc, al[HID + o], sb);
    }
    s1[(size_t)bh * N_ + n] = sa;
    s2[(size_t)bh * N_ + n] = sb;
}

// ---------------------------------------------------------------------------
// K2: layer-1 attention, one wave per (b,h, 16-row tile).
// ---------------------------------------------------------------------------
__global__ __launch_bounds__(32)
void k2_attn1(const float* __restrict__ adj, const _Float16* __restrict__ h1T,
              const float* __restrict__ s1g, const float* __restrict__ s2g,
              float* __restrict__ x1) {
    __shared__ float red[32];
    __shared__ float mrow[16];
    __shared__ float invl[16];
    __shared__ float s1row[16];

    const int TILES = N_ / 16;
    int tile = blockIdx.x % TILES;
    int bh   = blockIdx.x / TILES;          // b*NH + h
    int b    = bh / NH, h = bh % NH;
    int i0   = tile * 16;
    int L    = threadIdx.x;
    int r    = L & 15, hf = L >> 4;

    const float* adjRow = adj + (size_t)(i0 + r) * N_;
    const float* s2p    = s2g + (size_t)bh * N_;
    if (L < 16) s1row[L] = s1g[(size_t)bh * N_ + i0 + L];
    __syncthreads();
    float s1r  = s1row[r];
    int   irow = i0 + r;

    // ---- pass A: branchless masked row max (2 lanes per row) --------------
    float m = -INFINITY;
    for (int j0 = hf * 16; j0 < N_; j0 += 32) {
        float av[16], sv[16];
        load8f(adjRow + j0, av);     load8f(adjRow + j0 + 8, av + 8);
        load8f(s2p + j0, sv);        load8f(s2p + j0 + 8, sv + 8);
        #pragma unroll
        for (int t = 0; t < 16; ++t) {
            bool  valid = (av[t] > 0.f) || (j0 + t == irow);
            float z = s1r + sv[t];
            float e = z > 0.f ? z : LRELU_ALPHA * z;
            m = fmaxf(m, valid ? e : -INFINITY);
        }
    }
    red[L] = m; __syncthreads();
    if (L < 16) mrow[L] = fmaxf(red[L], red[L + 16]);
    __syncthreads();
    float mr = mrow[r];

    // ---- pass C: fused WMMA sweep + softmax denominator --------------------
    v8f acc0 = {}; v8f acc1 = {};
    float ls0 = 0.f, ls1 = 0.f;                 // two chains: shorter dep path
    const _Float16* hb = h1T + (size_t)bh * HID * N_;
    for (int jb = 0; jb < N_; jb += 32) {
        // A fragment: 16x32 P tile.  contiguous k order used here (B matches):
        // lanes 0-15 hold K 0-15, lanes 16-31 hold K 16-31.
        V16H af;
        #pragma unroll
        for (int g = 0; g < 2; ++g) {
            int jA = jb + hf * 16 + g * 8;
            float av[8], sv[8], pv[8];
            load8f(adjRow + jA, av);
            load8f(s2p + jA, sv);
            #pragma unroll
            for (int u = 0; u < 8; ++u) {
                bool  valid = (av[u] > 0.f) || (jA + u == irow);
                float z   = s1r + sv[u];
                float e   = z > 0.f ? z : LRELU_ALPHA * z;
                float arg = valid ? (e - mr) : -INFINITY;   // exp(-inf) = 0
                float p   = __expf(arg);
                pv[u] = p;
                if (u & 1) ls1 += p; else ls0 += p;
            }
            #pragma unroll
            for (int u = 0; u < 8; u += 2) {               // v_cvt_pk_rtz_f16_f32
                auto pk = __builtin_amdgcn_cvt_pkrtz(pv[u], pv[u + 1]);
                __builtin_memcpy(&af.h[g * 8 + u], &pk, 4);
            }
        }
        // B fragments: lane = n + 16*(k/16), half = k%16 -> contiguous in h1T
        int col = r;
        int j0  = jb + hf * 16;
        V16H bf0, bf1;
        const _Float16* p0 = hb + (size_t)col * N_ + j0;
        const _Float16* p1 = hb + (size_t)(col + 16) * N_ + j0;
        *(v8h*)&bf0.h[0] = *(const v8h*)p0;
        *(v8h*)&bf0.h[8] = *(const v8h*)(p0 + 8);
        *(v8h*)&bf1.h[0] = *(const v8h*)p1;
        *(v8h*)&bf1.h[8] = *(const v8h*)(p1 + 8);

        acc0 = __builtin_amdgcn_wmma_f32_16x16x32_f16(false, af.v, false, bf0.v,
                                                      (short)0, acc0, false, false);
        acc1 = __builtin_amdgcn_wmma_f32_16x16x32_f16(false, af.v, false, bf1.v,
                                                      (short)0, acc1, false, false);
    }

    // ---- denominator reduction (2 lanes per row) ---------------------------
    red[L] = ls0 + ls1; __syncthreads();
    if (L < 16) invl[L] = __builtin_amdgcn_rcpf(red[L] + red[L + 16]);
    __syncthreads();

    // ---- epilogue: C/D layout row = v + 8*hf, col = lane&15 ----------------
    V8F c0, c1; c0.v = acc0; c1.v = acc1;
    int col = r;
    #pragma unroll
    for (int vi = 0; vi < 8; ++vi) {
        int   row = vi + 8 * hf;
        float il  = invl[row];
        float y0  = elu_bl(c0.f[vi] * il);
        float y1  = elu_bl(c1.f[vi] * il);
        size_t ob = ((size_t)b * N_ + i0 + row) * CAT + h * HID;
        x1[ob + col]      = y0;
        x1[ob + 16 + col] = y1;
    }
}

// ---------------------------------------------------------------------------
// K3: layer-2 projections.  h2 = x1 @ W2 (K=192) -> h2T f16 [b][o][n], scores.
// ---------------------------------------------------------------------------
__global__ __launch_bounds__(256)
void k3_proj2(const float* __restrict__ x1, const float* __restrict__ W2,
              const float* __restrict__ a2, _Float16* __restrict__ h2T,
              float* __restrict__ s1, float* __restrict__ s2) {
    __shared__ float Wl[CAT * OUTC];
    __shared__ float al[2 * OUTC];
    int idx = blockIdx.x * 256 + threadIdx.x;   // b*N + n
    for (int t = threadIdx.x; t < CAT * OUTC; t += 256) Wl[t] = W2[t];
    if (threadIdx.x < 2 * OUTC) al[threadIdx.x] = a2[threadIdx.x];
    __syncthreads();

    int b = idx >> 11;
    int n = idx & (N_ - 1);
    const float* xp = x1 + (size_t)idx * CAT;
    float acc[OUTC];
    #pragma unroll
    for (int o = 0; o < OUTC; ++o) acc[o] = 0.f;
    for (int k = 0; k < CAT; ++k) {
        float xv = xp[k];
        #pragma unroll
        for (int o = 0; o < OUTC; ++o) acc[o] = fmaf(xv, Wl[k * OUTC + o], acc[o]);
    }
    float sa = 0.f, sb = 0.f;
    #pragma unroll
    for (int o = 0; o < OUTC; ++o) {
        h2T[((size_t)b * OUTC + o) * N_ + n] = (_Float16)acc[o];
        sa = fmaf(acc[o], al[o], sa);
        sb = fmaf(acc[o], al[OUTC + o], sb);
    }
    s1[idx] = sa;
    s2[idx] = sb;
}

// ---------------------------------------------------------------------------
// K4: layer-2 attention (O=16, single accumulator), ELU, final output.
// ---------------------------------------------------------------------------
__global__ __launch_bounds__(32)
void k4_attn2(const float* __restrict__ adj, const _Float16* __restrict__ h2T,
              const float* __restrict__ s1g, const float* __restrict__ s2g,
              float* __restrict__ out) {
    __shared__ float red[32];
    __shared__ float mrow[16];
    __shared__ float invl[16];
    __shared__ float s1row[16];

    const int TILES = N_ / 16;
    int tile = blockIdx.x % TILES;
    int b    = blockIdx.x / TILES;
    int i0   = tile * 16;
    int L    = threadIdx.x;
    int r    = L & 15, hf = L >> 4;

    const float* adjRow = adj + (size_t)(i0 + r) * N_;
    const float* s2p    = s2g + (size_t)b * N_;
    if (L < 16) s1row[L] = s1g[(size_t)b * N_ + i0 + L];
    __syncthreads();
    float s1r  = s1row[r];
    int   irow = i0 + r;

    float m = -INFINITY;
    for (int j0 = hf * 16; j0 < N_; j0 += 32) {
        float av[16], sv[16];
        load8f(adjRow + j0, av);     load8f(adjRow + j0 + 8, av + 8);
        load8f(s2p + j0, sv);        load8f(s2p + j0 + 8, sv + 8);
        #pragma unroll
        for (int t = 0; t < 16; ++t) {
            bool  valid = (av[t] > 0.f) || (j0 + t == irow);
            float z = s1r + sv[t];
            float e = z > 0.f ? z : LRELU_ALPHA * z;
            m = fmaxf(m, valid ? e : -INFINITY);
        }
    }
    red[L] = m; __syncthreads();
    if (L < 16) mrow[L] = fmaxf(red[L], red[L + 16]);
    __syncthreads();
    float mr = mrow[r];

    v8f acc = {};
    float ls0 = 0.f, ls1 = 0.f;
    const _Float16* hb = h2T + (size_t)b * OUTC * N_;
    for (int jb = 0; jb < N_; jb += 32) {
        V16H af;
        #pragma unroll
        for (int g = 0; g < 2; ++g) {
            int jA = jb + hf * 16 + g * 8;
            float av[8], sv[8], pv[8];
            load8f(adjRow + jA, av);
            load8f(s2p + jA, sv);
            #pragma unroll
            for (int u = 0; u < 8; ++u) {
                bool  valid = (av[u] > 0.f) || (jA + u == irow);
                float z   = s1r + sv[u];
                float e   = z > 0.f ? z : LRELU_ALPHA * z;
                float arg = valid ? (e - mr) : -INFINITY;
                float p   = __expf(arg);
                pv[u] = p;
                if (u & 1) ls1 += p; else ls0 += p;
            }
            #pragma unroll
            for (int u = 0; u < 8; u += 2) {
                auto pk = __builtin_amdgcn_cvt_pkrtz(pv[u], pv[u + 1]);
                __builtin_memcpy(&af.h[g * 8 + u], &pk, 4);
            }
        }
        int col = r;
        int j0  = jb + hf * 16;
        V16H bf;
        const _Float16* p0 = hb + (size_t)col * N_ + j0;
        *(v8h*)&bf.h[0] = *(const v8h*)p0;
        *(v8h*)&bf.h[8] = *(const v8h*)(p0 + 8);

        acc = __builtin_amdgcn_wmma_f32_16x16x32_f16(false, af.v, false, bf.v,
                                                     (short)0, acc, false, false);
    }

    red[L] = ls0 + ls1; __syncthreads();
    if (L < 16) invl[L] = __builtin_amdgcn_rcpf(red[L] + red[L + 16]);
    __syncthreads();

    V8F c; c.v = acc;
    int col = r;
    #pragma unroll
    for (int vi = 0; vi < 8; ++vi) {
        int   row = vi + 8 * hf;
        float y   = elu_bl(c.f[vi] * invl[row]);
        out[((size_t)b * N_ + i0 + row) * OUTC + col] = y;
    }
}

// ---------------------------------------------------------------------------
// Host-side launch.  Workspace layout (~10.2 MB, 256-B aligned chunks).
// ---------------------------------------------------------------------------
extern "C" void kernel_launch(void* const* d_in, const int* in_sizes, int n_in,
                              void* d_out, int out_size, void* d_ws, size_t ws_size,
                              hipStream_t stream) {
    (void)in_sizes; (void)n_in; (void)out_size; (void)ws_size;
    const float* x   = (const float*)d_in[0];
    const float* adj = (const float*)d_in[1];
    const float* W1  = (const float*)d_in[2];
    const float* a1  = (const float*)d_in[3];
    const float* W2  = (const float*)d_in[4];
    const float* a2  = (const float*)d_in[5];

    char*  ws  = (char*)d_ws;
    size_t off = 0;
    auto carve = [&](size_t bytes) -> char* {
        char* p = ws + off;
        off += (bytes + 255) & ~(size_t)255;
        return p;
    };
    _Float16* h1T = (_Float16*)carve(sizeof(_Float16) * (size_t)B_ * NH * HID * N_);
    float*    s1a = (float*)   carve(sizeof(float)    * (size_t)B_ * NH * N_);
    float*    s2a = (float*)   carve(sizeof(float)    * (size_t)B_ * NH * N_);
    float*    x1  = (float*)   carve(sizeof(float)    * (size_t)B_ * N_ * CAT);
    float*    s1b = (float*)   carve(sizeof(float)    * (size_t)B_ * N_);
    float*    s2b = (float*)   carve(sizeof(float)    * (size_t)B_ * N_);
    _Float16* h2T = (_Float16*)carve(sizeof(_Float16) * (size_t)B_ * OUTC * N_);

    k1_proj1<<<(B_ * NH * N_) / 256, 256, 0, stream>>>(x, W1, a1, h1T, s1a, s2a);
    k2_attn1<<<B_ * NH * (N_ / 16), 32, 0, stream>>>(adj, h1T, s1a, s2a, x1);
    k3_proj2<<<(B_ * N_) / 256, 256, 0, stream>>>(x1, W2, a2, h2T, s1b, s2b);
    k4_attn2<<<B_ * (N_ / 16), 32, 0, stream>>>(adj, h2T, s1b, s2b, (float*)d_out);
}
